// AttentionCircuit_35356170781339
// MI455X (gfx1250) — matrix operand order repaired and checked by reading, not compiled, not run on hardware
//
#include <hip/hip_runtime.h>
#include <hip/hip_bf16.h>

// ---------------------------------------------------------------------------
// Problem constants (from reference): B=4, S=1024, D=1024, H=16, dh=64,
// K_SEL=32, N_POOL=4096.
// ---------------------------------------------------------------------------
#define BATCH 4
#define SEQ   1024
#define DIM   1024
#define HEADS 16
#define DHEAD 64
#define KSEL  32

typedef __attribute__((ext_vector_type(16))) _Float16 v16h;
typedef __attribute__((ext_vector_type(8)))  float    v8f;
typedef __attribute__((ext_vector_type(4)))  _Float16 v4h;

__device__ __forceinline__ v8f wmma_f16(v16h a, v16h b, v8f c) {
  // D = A(16x32,f16) * B(32x16,f16) + C(16x16,f32)
  return __builtin_amdgcn_wmma_f32_16x16x32_f16(
      /*neg_a=*/false, a, /*neg_b=*/false, b,
      /*c_mod=*/(short)0, c, /*reuse_a=*/false, /*reuse_b=*/false);
}

// ---- DPP16 butterfly reductions over a 16-lane row (VALU, no DS traffic) ---
// Sequence quad_perm(1,0,3,2)=0xB1, quad_perm(2,3,0,1)=0x4E,
// ROW_HALF_MIRROR=0x141, ROW_MIRROR=0x140 leaves every lane of each 16-lane
// row holding the full row reduction (values are quad-uniform at each stage).
template <int CTRL>
__device__ __forceinline__ float dpp_f32(float x) {
  return __int_as_float(__builtin_amdgcn_update_dpp(
      0, __float_as_int(x), CTRL, 0xf, 0xf, true));
}
__device__ __forceinline__ float row16_max(float x) {
  x = fmaxf(x, dpp_f32<0xB1>(x));
  x = fmaxf(x, dpp_f32<0x4E>(x));
  x = fmaxf(x, dpp_f32<0x141>(x));
  x = fmaxf(x, dpp_f32<0x140>(x));
  return x;
}
__device__ __forceinline__ float row16_sum(float x) {
  x += dpp_f32<0xB1>(x);
  x += dpp_f32<0x4E>(x);
  x += dpp_f32<0x141>(x);
  x += dpp_f32<0x140>(x);
  return x;
}

// A fragment: A[m][k], m = lane&15. lanes<16: K in {0..15};
// lanes>=16: +16.  tile is row-major with leading dim `ld` (halves).
__device__ __forceinline__ v16h load_a_frag(const _Float16* tile, int ld) {
  const int lane = threadIdx.x & 31;
  const int m    = lane & 15;
  const int koff = (lane >> 4) << 4;       // 0 or 16
  v16h a;
#pragma unroll
  for (int j = 0; j < 8; ++j) {
    const int kb = 2 * j;
    a[2 * j]     = tile[m * ld + kb + koff + 0];
    a[2 * j + 1] = tile[m * ld + kb + koff + 1];
  }
  return a;
}

// B fragment where B[k][n] = src[k*ld + n]  (B stored row-major, 32x16 slice)
__device__ __forceinline__ v16h load_b_frag(const _Float16* src, int ld) {
  const int lane = threadIdx.x & 31;
  const int n    = lane & 15;
  const int koff = (lane >> 4) << 4;       // 0 or 16
  v16h b;
#pragma unroll
  for (int h = 0; h < 16; ++h) b[h] = src[(koff + h) * ld + n];
  return b;
}

// B fragment where B[k][n] = src[n*ld + k]  (src holds B^T row-major;
// used for Q·K^T where K rows are [key][d]).
__device__ __forceinline__ v16h load_bT_frag(const _Float16* src, int ld) {
  const int lane = threadIdx.x & 31;
  const int n    = lane & 15;
  const int koff = (lane >> 4) << 4;       // 0 or 16
  v16h b;
#pragma unroll
  for (int h = 0; h < 16; ++h) b[h] = src[n * ld + koff + h];
  return b;
}

// ---------------------------------------------------------------------------
// Kernel 1: per-token QKV construction (gathers are L2-resident: pools are
// 16 MB each << 192 MB L2). fp32 math, f16 output for the WMMA stages.
// One block (256 threads = 8 waves) per token.
// ---------------------------------------------------------------------------
__global__ __launch_bounds__(256)
void qkv_build_kernel(const float* __restrict__ x,
                      const float* __restrict__ gQ,
                      const float* __restrict__ gK,
                      const float* __restrict__ gV,
                      const int*   __restrict__ ci_qk,
                      const int*   __restrict__ ci_v,
                      const float* __restrict__ qkn,
                      const float* __restrict__ vn,
                      _Float16* __restrict__ Qh,
                      _Float16* __restrict__ Kh,
                      _Float16* __restrict__ Vh) {
  const int token = blockIdx.x;            // b*S + s
  const int tid   = threadIdx.x;
  const int wv    = tid >> 5;
  const int lane  = tid & 31;

  __shared__ __align__(16) float xs[DIM];
  __shared__ float act_qk[KSEL];
  __shared__ float act_v[KSEL];

  // vectorized x load: 256 float4's, one per thread
  ((float4*)xs)[tid] = ((const float4*)(x + (size_t)token * DIM))[tid];
  __syncthreads();

  // Each of the 8 waves computes 4 qk dots + 4 v dots (32 each total).
#pragma unroll
  for (int u = 0; u < 4; ++u) {
    const int i = wv * 4 + u;
    {
      const int r = ci_qk[(size_t)token * KSEL + i];
      const float* row = qkn + (size_t)r * DIM;
      float s = 0.f;
#pragma unroll 4
      for (int d = lane; d < DIM; d += 32) s += xs[d] * row[d];
      s = row16_sum(s);
      s += __shfl_xor(s, 16, 32);          // combine the two 16-lane halves
      if (lane == 0) act_qk[i] = s;
    }
    {
      const int r = ci_v[(size_t)token * KSEL + i];
      const float* row = vn + (size_t)r * DIM;
      float s = 0.f;
#pragma unroll 4
      for (int d = lane; d < DIM; d += 32) s += xs[d] * row[d];
      s = row16_sum(s);
      s += __shfl_xor(s, 16, 32);
      if (lane == 0) act_v[i] = s;
    }
  }
  __syncthreads();

  // Gated reconstruction: thread owns 4 consecutive d's.
  const int d0 = tid * 4;
  float aq[4] = {0, 0, 0, 0}, ak[4] = {0, 0, 0, 0}, av[4] = {0, 0, 0, 0};
#pragma unroll 4
  for (int i = 0; i < KSEL; ++i) {
    const float a  = act_qk[i];
    const float wq = a * gQ[(size_t)token * KSEL + i];
    const float wk = a * gK[(size_t)token * KSEL + i];
    const int   r  = ci_qk[(size_t)token * KSEL + i];
    const float4 c = *(const float4*)(qkn + (size_t)r * DIM + d0);
    aq[0] += wq * c.x; aq[1] += wq * c.y; aq[2] += wq * c.z; aq[3] += wq * c.w;
    ak[0] += wk * c.x; ak[1] += wk * c.y; ak[2] += wk * c.z; ak[3] += wk * c.w;
  }
#pragma unroll 4
  for (int i = 0; i < KSEL; ++i) {
    const float wvv = act_v[i] * gV[(size_t)token * KSEL + i];
    const int   r   = ci_v[(size_t)token * KSEL + i];
    const float4 c  = *(const float4*)(vn + (size_t)r * DIM + d0);
    av[0] += wvv * c.x; av[1] += wvv * c.y; av[2] += wvv * c.z; av[3] += wvv * c.w;
  }
  const size_t base = (size_t)token * DIM + d0;
  v4h pq = {(_Float16)aq[0], (_Float16)aq[1], (_Float16)aq[2], (_Float16)aq[3]};
  v4h pk = {(_Float16)ak[0], (_Float16)ak[1], (_Float16)ak[2], (_Float16)ak[3]};
  v4h pv = {(_Float16)av[0], (_Float16)av[1], (_Float16)av[2], (_Float16)av[3]};
  *(v4h*)(Qh + base) = pq;   // 8-byte packed stores
  *(v4h*)(Kh + base) = pk;
  *(v4h*)(Vh + base) = pv;
}

// ---------------------------------------------------------------------------
// Kernel 2: causal flash attention, f16 WMMA, f32 accumulate.
// grid = (S/64, B*H); block = 128 (4 waves). Wave w owns q-rows [w*16, w*16+16).
// Key blocks of 32, online softmax with DPP row reductions.
// ---------------------------------------------------------------------------
__global__ __launch_bounds__(128)
void flash_attn_kernel(const _Float16* __restrict__ Qh,
                       const _Float16* __restrict__ Kh,
                       const _Float16* __restrict__ Vh,
                       _Float16* __restrict__ Ah) {
  __shared__ __align__(16) _Float16 Qs[64 * DHEAD];
  __shared__ __align__(16) _Float16 Ks[32 * DHEAD];
  __shared__ __align__(16) _Float16 Vs[32 * DHEAD];
  __shared__ __align__(16) _Float16 Ps[4 * 16 * 32]; // per-wave P bounce

  const int tid  = threadIdx.x;
  const int lane = tid & 31;
  const int wv   = tid >> 5;
  const int nlo  = lane & 15;
  const int hi   = lane >> 4;              // 0 or 1
  const int qb   = blockIdx.x;             // 64-row query block
  const int bh   = blockIdx.y;
  const int b    = bh >> 4;                // H = 16
  const int h    = bh & 15;

  const size_t base = (size_t)b * SEQ * DIM + (size_t)h * DHEAD;
  const _Float16* Qg = Qh + base + (size_t)(qb * 64) * DIM;

  // Q tile: 64x64 halves = 512 uint4's; 4 per thread, 16B-aligned.
#pragma unroll
  for (int e = tid; e < 512; e += 128) {
    const int r = e >> 3, c8 = e & 7;      // 8 uint4 per 64-half row
    *(uint4*)&Qs[r * DHEAD + c8 * 8] =
        *(const uint4*)&Qg[(size_t)r * DIM + c8 * 8];
  }

  float m[8], l[8];
  v8f o[4];
  const v8f vzero = {0.f, 0.f, 0.f, 0.f, 0.f, 0.f, 0.f, 0.f};
#pragma unroll
  for (int r = 0; r < 8; ++r) { m[r] = -1e30f; l[r] = 0.f; }
#pragma unroll
  for (int c = 0; c < 4; ++c) o[c] = vzero;

  const int qrow0 = qb * 64 + wv * 16 + hi * 8;  // row of o[*][r] is qrow0 + r
  const int nkb   = 2 * qb + 2;                  // causal: keys up to qb*64+63

  for (int kb = 0; kb < nkb; ++kb) {
    __syncthreads();                       // all PV reads of Vs done
    const _Float16* Kg = Kh + base + (size_t)(kb * 32) * DIM;
    const _Float16* Vg = Vh + base + (size_t)(kb * 32) * DIM;
    // K/V tiles: 32x64 halves = 256 uint4's each; 2 per thread.
#pragma unroll
    for (int e = tid; e < 256; e += 128) {
      const int r = e >> 3, c8 = e & 7;
      *(uint4*)&Ks[r * DHEAD + c8 * 8] =
          *(const uint4*)&Kg[(size_t)r * DIM + c8 * 8];
      *(uint4*)&Vs[r * DHEAD + c8 * 8] =
          *(const uint4*)&Vg[(size_t)r * DIM + c8 * 8];
    }
    if (kb + 1 < nkb) {  // warm caches for next tile (global_prefetch_b8)
      __builtin_prefetch(Kg + (size_t)32 * DIM + tid * 16, 0, 1);
      __builtin_prefetch(Vg + (size_t)32 * DIM + tid * 16, 0, 1);
    }
    __syncthreads();

    // S = Q(16x64) · K^T(64x32) -> two 16x16 C-tiles (keys lo/hi 16)
    v8f s0 = vzero, s1 = vzero;
#pragma unroll
    for (int kc = 0; kc < 2; ++kc) {
      const v16h aq = load_a_frag(&Qs[(wv * 16) * DHEAD + kc * 32], DHEAD);
      const v16h b0 = load_bT_frag(&Ks[0 * DHEAD + kc * 32], DHEAD);
      const v16h b1 = load_bT_frag(&Ks[16 * DHEAD + kc * 32], DHEAD);
      s0 = wmma_f16(aq, b0, s0);
      s1 = wmma_f16(aq, b1, s1);
    }

    // Online softmax (scale = 1/sqrt(64) = 0.125). Row stats via DPP16
    // butterflies (wave32: lanes<16 are rows 0..7, lanes>=16 rows 8..15).
    const int col0 = kb * 32 + nlo;
    const int col1 = col0 + 16;
    float p0[8], p1[8], fs[8];
#pragma unroll
    for (int r = 0; r < 8; ++r) {
      const int row = qrow0 + r;
      float v0 = s0[r] * 0.125f;
      float v1 = s1[r] * 0.125f;
      if (col0 > row) v0 = -1e30f;
      if (col1 > row) v1 = -1e30f;
      const float mx = row16_max(fmaxf(v0, v1));
      const float mn = fmaxf(m[r], mx);
      const float e0 = __expf(v0 - mn);
      const float e1 = __expf(v1 - mn);
      const float rs = row16_sum(e0 + e1);
      const float f  = __expf(m[r] - mn);
      l[r] = l[r] * f + rs;
      m[r] = mn;
      fs[r] = f;
      p0[r] = e0;
      p1[r] = e1;
    }
#pragma unroll
    for (int c = 0; c < 4; ++c)
#pragma unroll
      for (int r = 0; r < 8; ++r) o[c][r] *= fs[r];

    // Re-lay P (C/D layout) into an A fragment via a *wave-private* LDS
    // bounce. Wave32 lanes are lockstep and LDS ops are in-order per wave,
    // so a split-counter wait suffices -- no block barrier needed.
    _Float16* Pw = &Ps[wv * 16 * 32];
#pragma unroll
    for (int r = 0; r < 8; ++r) {
      Pw[(hi * 8 + r) * 32 + nlo]      = (_Float16)p0[r];
      Pw[(hi * 8 + r) * 32 + nlo + 16] = (_Float16)p1[r];
    }
    asm volatile("s_wait_dscnt 0x0" ::: "memory");

    const v16h ap = load_a_frag(Pw, 32);
#pragma unroll
    for (int c = 0; c < 4; ++c) {
      const v16h bv = load_b_frag(&Vs[c * 16], DHEAD);
      o[c] = wmma_f16(ap, bv, o[c]);
    }
  }

  // Normalize and store (f16 for the W_O GEMM stage).
  float inv[8];
#pragma unroll
  for (int r = 0; r < 8; ++r) inv[r] = 1.f / l[r];
#pragma unroll
  for (int c = 0; c < 4; ++c)
#pragma unroll
    for (int r = 0; r < 8; ++r) {
      Ah[base + (size_t)(qrow0 + r) * DIM + c * 16 + nlo] =
          (_Float16)(o[c][r] * inv[r]);
    }
}

// ---------------------------------------------------------------------------
// Kernel 3: out = Ah(4096x1024,f16) · W_O(1024x1024,f32->f16 in LDS), fp32 out.
// grid = (M/64, N/64); block = 128 (4 waves). Wave w: rows w*16..w*16+15.
// W_O is 4 MB -> fully L2-resident across the grid.
// ---------------------------------------------------------------------------
__global__ __launch_bounds__(128)
void gemm_wo_kernel(const _Float16* __restrict__ A,
                    const float* __restrict__ W,
                    float* __restrict__ C) {
  __shared__ __align__(16) _Float16 As[64 * 32];
  __shared__ __align__(16) _Float16 Ws[32 * 64];

  const int tid  = threadIdx.x;
  const int lane = tid & 31;
  const int wv   = tid >> 5;
  const int nlo  = lane & 15;
  const int hi   = lane >> 4;
  const int mb   = blockIdx.x;
  const int nb   = blockIdx.y;

  const v8f vzero = {0.f, 0.f, 0.f, 0.f, 0.f, 0.f, 0.f, 0.f};
  v8f acc[4];
#pragma unroll
  for (int c = 0; c < 4; ++c) acc[c] = vzero;

  for (int kk = 0; kk < DIM; kk += 32) {
    __syncthreads();
    // A tile: 64x32 halves = 256 uint4's; 2 per thread.
#pragma unroll
    for (int e = tid; e < 256; e += 128) {
      const int r = e >> 2, c8 = e & 3;    // 4 uint4 per 32-half row
      *(uint4*)&As[r * 32 + c8 * 8] =
          *(const uint4*)&A[(size_t)(mb * 64 + r) * DIM + kk + c8 * 8];
    }
    // W tile: 32x64 floats -> f16; float4 loads, 8-byte packed f16 stores.
#pragma unroll
    for (int e = tid; e < 512; e += 128) {
      const int r = e >> 4, c4 = e & 15;   // 16 float4 per 64-col row
      const float4 w = *(const float4*)&W[(size_t)(kk + r) * DIM + nb * 64 + c4 * 4];
      v4h wh = {(_Float16)w.x, (_Float16)w.y, (_Float16)w.z, (_Float16)w.w};
      *(v4h*)&Ws[r * 64 + c4 * 4] = wh;
    }
    if (kk + 32 < DIM) {
      __builtin_prefetch(&A[(size_t)(mb * 64 + (tid >> 1)) * DIM + kk + 32], 0, 1);
      __builtin_prefetch(&W[(size_t)(kk + 32 + (tid >> 2)) * DIM + nb * 64], 0, 1);
    }
    __syncthreads();

    const v16h af = load_a_frag(&As[wv * 16 * 32], 32);
#pragma unroll
    for (int c4 = 0; c4 < 4; ++c4) {
      const v16h bf = load_b_frag(&Ws[c4 * 16], 64);
      acc[c4] = wmma_f16(af, bf, acc[c4]);
    }
  }

  const int row0 = mb * 64 + wv * 16 + hi * 8;
#pragma unroll
  for (int c4 = 0; c4 < 4; ++c4)
#pragma unroll
    for (int r = 0; r < 8; ++r)
      C[(size_t)(row0 + r) * DIM + nb * 64 + c4 * 16 + nlo] = acc[c4][r];
}

// ---------------------------------------------------------------------------
extern "C" void kernel_launch(void* const* d_in, const int* in_sizes, int n_in,
                              void* d_out, int out_size, void* d_ws, size_t ws_size,
                              hipStream_t stream) {
  const float* x     = (const float*)d_in[0];
  const float* g_Q   = (const float*)d_in[1];
  const float* g_K   = (const float*)d_in[2];
  const float* g_V   = (const float*)d_in[3];
  const int*   ci_qk = (const int*)d_in[4];
  const int*   ci_v  = (const int*)d_in[5];
  const float* qkn   = (const float*)d_in[6];
  const float* vnrn  = (const float*)d_in[7];
  const float* W_O   = (const float*)d_in[8];

  // Workspace: four f16 [B*S, D] buffers (8 MB each, 32 MB total).
  const size_t bufBytes = (size_t)BATCH * SEQ * DIM * sizeof(_Float16);
  char* ws = (char*)d_ws;
  _Float16* Qh = (_Float16*)(ws + 0 * bufBytes);
  _Float16* Kh = (_Float16*)(ws + 1 * bufBytes);
  _Float16* Vh = (_Float16*)(ws + 2 * bufBytes);
  _Float16* Ah = (_Float16*)(ws + 3 * bufBytes);

  // Stage 1: QKV construction (one block per token).
  qkv_build_kernel<<<BATCH * SEQ, 256, 0, stream>>>(
      x, g_Q, g_K, g_V, ci_qk, ci_v, qkn, vnrn, Qh, Kh, Vh);

  // Stage 2: causal flash attention.
  flash_attn_kernel<<<dim3(SEQ / 64, BATCH * HEADS), 128, 0, stream>>>(
      Qh, Kh, Vh, Ah);

  // Stage 3: output projection.
  gemm_wo_kernel<<<dim3(BATCH * SEQ / 64, DIM / 64), 128, 0, stream>>>(
      Ah, W_O, (float*)d_out);
}
